// Self_Attn_3D_27247272526194
// MI455X (gfx1250) — matrix operand order repaired and checked by reading, not compiled, not run on hardware
//
#include <hip/hip_runtime.h>
#include <hip/hip_bf16.h>

// Self-attention (B=8, N=4096, Cin=256, Cqk=32) for gfx1250 / MI455X.
// Flash-attention style; all matmuls via v_wmma_f32_16x16x32_bf16.
// K/V tiles staged in LDS (V via Tensor Data Mover), double buffered.

#define NTOK 4096
#define CIN  256
#define CQK  32
#define NJT  (NTOK / 32)   // 128 key-tile steps

typedef __attribute__((ext_vector_type(16))) __bf16 v16bf;
typedef __attribute__((ext_vector_type(8)))  float  v8f;
typedef __attribute__((ext_vector_type(4)))  unsigned int u32x4;
typedef __attribute__((ext_vector_type(8)))  int i32x8;
typedef __attribute__((ext_vector_type(4)))  int i32x4;

union ABfrag { v16bf bf; unsigned int u[8]; uint4 q[2]; };

#if defined(__has_builtin)
#  if __has_builtin(__builtin_amdgcn_tensor_load_to_lds) && \
      __has_builtin(__builtin_amdgcn_s_wait_tensorcnt)
#    define USE_TDM 1
#  endif
#endif
#ifndef USE_TDM
#  define USE_TDM 0
#endif

__device__ inline unsigned short f2bf(float f) {
    unsigned int u = __float_as_uint(f);
    u += 0x7FFFu + ((u >> 16) & 1u);          // round-to-nearest-even
    return (unsigned short)(u >> 16);
}
__device__ inline unsigned int pack2bf(float lo, float hi) {
    return (unsigned int)f2bf(lo) | ((unsigned int)f2bf(hi) << 16);
}
__device__ inline v8f wmma_bf16(v16bf a, v16bf b, v8f c) {
    return __builtin_amdgcn_wmma_f32_16x16x32_bf16(
        false, a, false, b, (short)0, c, false, false);
}

// ---------------------------------------------------------------------------
// Kernel 0a: x -> bf16 row-major copy + per-token channel mean (f32 accuracy).
// One wave per token, fully coalesced.
// ---------------------------------------------------------------------------
__global__ __launch_bounds__(256) void cvt_x_kernel(
    const float* __restrict__ x, unsigned short* __restrict__ xb,
    float* __restrict__ xmean)
{
    const int lane = threadIdx.x & 31;
    const int wave = threadIdx.x >> 5;
    const int tok  = blockIdx.x * 8 + wave;
    const float* xr = x + (size_t)tok * CIN;
    unsigned short* orow = xb + (size_t)tok * CIN;
    float s = 0.f;
    #pragma unroll
    for (int i = 0; i < CIN; i += 32) {
        float v = xr[i + lane];
        s += v;
        orow[i + lane] = f2bf(v);
    }
    #pragma unroll
    for (int off = 16; off; off >>= 1) s += __shfl_xor(s, off, 32);
    if (lane == 0) xmean[tok] = s * (1.0f / (float)CIN);
}

// ---------------------------------------------------------------------------
// Kernel 0b: weights -> bf16, transposed to (out_ch, in_ch) so that B-frags
// become two contiguous 16B loads (k runs along rows).
// ---------------------------------------------------------------------------
__global__ __launch_bounds__(256) void cvt_w_kernel(
    const float* __restrict__ Wq, const float* __restrict__ Wk,
    const float* __restrict__ Wv,
    unsigned short* __restrict__ Wqt, unsigned short* __restrict__ Wkt,
    unsigned short* __restrict__ Wvt)
{
    int idx = blockIdx.x * 256 + threadIdx.x;
    if (idx < CIN * CIN) {                  // Wv: (k,n) -> Wvt[n*256+k]
        int k = idx >> 8, n = idx & 255;
        Wvt[n * CIN + k] = f2bf(Wv[idx]);
    }
    if (idx < CIN * CQK) {                  // Wq/Wk: (k,n) -> Wt[n*256+k]
        int k = idx >> 5, n = idx & 31;
        Wqt[n * CIN + k] = f2bf(Wq[idx]);
        Wkt[n * CIN + k] = f2bf(Wk[idx]);
    }
}

// ---------------------------------------------------------------------------
// Kernel 1: projections. Each wave: 16-token tile.
//   Qb,Kb : (B*N, 32) bf16 row-major; Vt : (B, 256, N) bf16 channel-major.
// All fragments are pure vector loads (pre-converted bf16 operands).
// ---------------------------------------------------------------------------
__global__ __launch_bounds__(256) void proj_kernel(
    const unsigned short* __restrict__ xb,
    const unsigned short* __restrict__ Wqt, const float* __restrict__ bq,
    const unsigned short* __restrict__ Wkt, const float* __restrict__ bk,
    const unsigned short* __restrict__ Wvt, const float* __restrict__ bv,
    unsigned short* __restrict__ Qb, unsigned short* __restrict__ Kb,
    unsigned short* __restrict__ Vt)
{
    const int lane = threadIdx.x & 31;
    const int wave = threadIdx.x >> 5;
    const int m = lane & 15;
    const int h = lane >> 4;
    const int t0 = (blockIdx.x * 8 + wave) * 16;
    const int b  = t0 >> 12;
    const int nloc = t0 & (NTOK - 1);

    // ---- Q and K projections (4 output tiles) ----
    {
        v8f acc[4] = {};
        for (int kc = 0; kc < 8; ++kc) {
            ABfrag xa;
            const unsigned short* xr = xb + (size_t)(t0 + m) * CIN + kc * 32 + 8 * h;
            xa.q[0] = *(const uint4*)xr;
            xa.q[1] = *(const uint4*)(xr + 16);
            #pragma unroll
            for (int t = 0; t < 4; ++t) {
                const unsigned short* Wt = (t < 2) ? Wqt : Wkt;
                int n0 = (t & 1) * 16;
                ABfrag wb;
                const unsigned short* wr = Wt + (size_t)(n0 + m) * CIN + kc * 32 + 8 * h;
                wb.q[0] = *(const uint4*)wr;
                wb.q[1] = *(const uint4*)(wr + 16);
                acc[t] = wmma_bf16(xa.bf, wb.bf, acc[t]);
            }
        }
        #pragma unroll
        for (int t = 0; t < 4; ++t) {
            const float* bias = (t < 2) ? bq : bk;
            unsigned short* Ob = (t < 2) ? Qb : Kb;
            int n0 = (t & 1) * 16;
            float bsv = bias[n0 + m];
            #pragma unroll
            for (int r = 0; r < 8; ++r) {
                int tok = t0 + r + 8 * h;
                Ob[(size_t)tok * CQK + n0 + m] = f2bf(acc[t][r] + bsv);
            }
        }
    }

    // ---- V projection, 4 groups of 4 tiles, stored channel-major ----
    for (int vg = 0; vg < 4; ++vg) {
        v8f acc[4] = {};
        for (int kc = 0; kc < 8; ++kc) {
            ABfrag xa;
            const unsigned short* xr = xb + (size_t)(t0 + m) * CIN + kc * 32 + 8 * h;
            xa.q[0] = *(const uint4*)xr;
            xa.q[1] = *(const uint4*)(xr + 16);
            #pragma unroll
            for (int t = 0; t < 4; ++t) {
                int n0 = (vg * 4 + t) * 16;
                ABfrag wb;
                const unsigned short* wr = Wvt + (size_t)(n0 + m) * CIN + kc * 32 + 8 * h;
                wb.q[0] = *(const uint4*)wr;
                wb.q[1] = *(const uint4*)(wr + 16);
                acc[t] = wmma_bf16(xa.bf, wb.bf, acc[t]);
            }
        }
        #pragma unroll
        for (int t = 0; t < 4; ++t) {
            int c = (vg * 4 + t) * 16 + m;
            float bsv = bv[c];
            unsigned short* vrow = Vt + ((size_t)b * CIN + c) * NTOK;
            #pragma unroll
            for (int r = 0; r < 8; ++r)
                vrow[nloc + r + 8 * h] = f2bf(acc[t][r] + bsv);
        }
    }
}

// ---------------------------------------------------------------------------
// Kernel 2: streaming attention. Block = 8 waves x 16 queries, all sharing
// LDS-staged K (cooperative) and V (Tensor Data Mover) tiles, double buffered.
// ---------------------------------------------------------------------------
__global__ __launch_bounds__(256) void attn_kernel(
    const unsigned short* __restrict__ Qb, const unsigned short* __restrict__ Kb,
    const unsigned short* __restrict__ Vt, const float* __restrict__ xmean,
    const float* __restrict__ gamma, float* __restrict__ out)
{
    __shared__ unsigned short Klds[2][32 * CQK];    // 2 x 2 KB
    __shared__ unsigned short Vlds[2][32 * CIN];    // 2 x 16 KB (ch-major, 32 keys/row)

    const int lane = threadIdx.x & 31;
    const int wave = threadIdx.x >> 5;
    const int m = lane & 15;
    const int h = lane >> 4;
    const int t0 = (blockIdx.x * 8 + wave) * 16;    // query tile (global tokens)
    const int b  = t0 >> 12;
    const int keybase = b << 12;

    // fixed Q^T B-fragment: lane -> query, VGPRs -> channel
    ABfrag qf;
    {
        const unsigned short* qr = Qb + (size_t)(t0 + m) * CQK + 8 * h;
        qf.q[0] = *(const uint4*)qr;
        qf.q[1] = *(const uint4*)(qr + 16);
    }

#if USE_TDM
    unsigned vofs0 = (unsigned)(uintptr_t)&Vlds[0][0];
    unsigned vofs1 = (unsigned)(uintptr_t)&Vlds[1][0];
#endif

    // cooperative K tile stage: 2 KB contiguous -> 128 uint4
    auto stage_k = [&](int buf, int j) {
        if (threadIdx.x < 128) {
            const uint4* src = (const uint4*)(Kb + (size_t)(keybase + j) * CQK);
            ((uint4*)&Klds[buf][0])[threadIdx.x] = src[threadIdx.x];
        }
    };
#if USE_TDM
    // TDM 2-D tile: dim0 = 32 keys (contiguous), dim1 = 256 channels, stride 4096
    auto tdm_v = [&](int buf, int j) {
        unsigned long long ga =
            (unsigned long long)(uintptr_t)(Vt + ((size_t)b * CIN) * NTOK + j);
        u32x4 g0;
        g0[0] = 1u;                                      // count=1 (valid user D#)
        g0[1] = buf ? vofs1 : vofs0;                     // lds_addr (bytes)
        g0[2] = (unsigned)ga;                            // global_addr[31:0]
        g0[3] = (unsigned)((ga >> 32) & 0x1FFFFFFu) | (2u << 30);  // [56:32]|type=2
        i32x8 g1;
        g1[0] = (int)(1u << 16);                         // data_size = 2 bytes
        g1[1] = (int)((unsigned)NTOK << 16);             // tensor_dim0 = 4096 (lo16)
        g1[2] = (int)((unsigned)CIN << 16);              // dim0 hi16=0 | tensor_dim1=256
        g1[3] = (int)(32u << 16);                        // dim1 hi16=0 | tile_dim0=32
        g1[4] = (int)(unsigned)CIN;                      // tile_dim1=256, tile_dim2=0
        g1[5] = (int)(unsigned)NTOK;                     // tensor_dim0_stride = 4096
        g1[6] = 0;
        g1[7] = 0;
        i32x4 z4 = {0, 0, 0, 0};                         // groups 2/3: NULL (2-D)
        i32x8 z8 = {0, 0, 0, 0, 0, 0, 0, 0};             // trailing group (clang-23 ABI)
        __builtin_amdgcn_tensor_load_to_lds(g0, g1, z4, z4, z8, 0);
    };
#else
    // fallback: cooperative V tile stage, one 64B row per thread
    auto stage_v = [&](int buf, int j) {
        const uint4* src =
            (const uint4*)(Vt + ((size_t)b * CIN + threadIdx.x) * NTOK + j);
        uint4* dst = (uint4*)&Vlds[buf][threadIdx.x * 32];
        #pragma unroll
        for (int i = 0; i < 4; ++i) dst[i] = src[i];
    };
#endif

    // transposed score tiles from LDS K: T = K_tile(16x32) x Q^T(32x16)
    auto qk_tiles = [&](int buf, v8f& T1, v8f& T2) {
        ABfrag k1, k2;
        const unsigned short* kr1 = &Klds[buf][m * CQK];
        const unsigned short* kr2 = kr1 + 16 * CQK;
        k1.q[0] = *(const uint4*)(kr1 + 8 * h);
        k1.q[1] = *(const uint4*)(kr1 + 16 + 8 * h);
        k2.q[0] = *(const uint4*)(kr2 + 8 * h);
        k2.q[1] = *(const uint4*)(kr2 + 16 + 8 * h);
        v8f z = {};
        T1 = wmma_bf16(k1.bf, qf.bf, z);
        T2 = wmma_bf16(k2.bf, qf.bf, z);
    };

    // ---- pass 1: online max & sum over all keys (lane-local in T layout) ----
    float mx = -3.0e38f, l = 0.f;
    stage_k(0, 0);
    for (int jt = 0; jt < NJT; ++jt) {
        int cur = jt & 1;
        __syncthreads();
        if (jt + 1 < NJT) stage_k(cur ^ 1, (jt + 1) * 32);
        v8f T1, T2;
        qk_tiles(cur, T1, T2);
        float tmax = T1[0];
        #pragma unroll
        for (int r = 1; r < 8; ++r) tmax = fmaxf(tmax, T1[r]);
        #pragma unroll
        for (int r = 0; r < 8; ++r) tmax = fmaxf(tmax, T2[r]);
        float mn = fmaxf(mx, tmax);
        float s = l * __expf(mx - mn);
        #pragma unroll
        for (int r = 0; r < 8; ++r)
            s += __expf(T1[r] - mn) + __expf(T2[r] - mn);
        mx = mn; l = s;
    }
    { // combine the two half-wave key partitions
        float mo = __shfl_xor(mx, 16, 32);
        float lo = __shfl_xor(l, 16, 32);
        float mt = fmaxf(mx, mo);
        l = l * __expf(mx - mt) + lo * __expf(mo - mt);
        mx = mt;
    }
    float rl = 1.0f / l;
    float rlb[8];
    #pragma unroll
    for (int r = 0; r < 8; ++r) rlb[r] = __shfl(rl, r + 8 * h, 32);

    // ---- pass 2: P·V over all 256 channels, V tiles via TDM double buffer ----
    v8f acc[16] = {};
    __syncthreads();                 // pass-1 consumers done before restaging
    stage_k(0, 0);
#if USE_TDM
    if (wave == 0) tdm_v(0, 0);
#else
    stage_v(0, 0);
#endif
    for (int jt = 0; jt < NJT; ++jt) {
        int cur = jt & 1, nxt = cur ^ 1;
#if USE_TDM
        if (wave == 0) __builtin_amdgcn_s_wait_tensorcnt(0);
#endif
        __syncthreads();             // cur tiles visible; nxt buffers free
        if (jt + 1 < NJT) {
            stage_k(nxt, (jt + 1) * 32);
#if USE_TDM
            if (wave == 0) tdm_v(nxt, (jt + 1) * 32);
#else
            stage_v(nxt, (jt + 1) * 32);
#endif
        }
        v8f T1, T2;
        qk_tiles(cur, T1, T2);
        // P fragment: transposed C tiles reinterpreted as bf16 A-frag (lane-local)
        ABfrag pa;
        #pragma unroll
        for (int v = 0; v < 4; ++v)
            pa.u[v] = pack2bf(__expf(T1[2 * v] - mx), __expf(T1[2 * v + 1] - mx));
        #pragma unroll
        for (int v = 0; v < 4; ++v)
            pa.u[4 + v] = pack2bf(__expf(T2[2 * v] - mx), __expf(T2[2 * v + 1] - mx));
        #pragma unroll
        for (int ct = 0; ct < 16; ++ct) {
            ABfrag vf;
            const unsigned short* vr = &Vlds[cur][(ct * 16 + m) * 32 + 8 * h];
            vf.q[0] = *(const uint4*)vr;
            vf.q[1] = *(const uint4*)(vr + 16);
            acc[ct] = wmma_bf16(pa.bf, vf.bf, acc[ct]);
        }
    }

    // ---- epilogue: out = gamma/256 * sum_c(attn_out/l) + mean_c(x) ----
    float csum[8] = {0.f, 0.f, 0.f, 0.f, 0.f, 0.f, 0.f, 0.f};
    #pragma unroll
    for (int ct = 0; ct < 16; ++ct)
        #pragma unroll
        for (int r = 0; r < 8; ++r) csum[r] += acc[ct][r];
    float g = gamma[0] * (1.0f / (float)CIN);
    #pragma unroll
    for (int r = 0; r < 8; ++r) {
        float v = csum[r] * rlb[r];
        #pragma unroll
        for (int off = 1; off < 16; off <<= 1) v += __shfl_xor(v, off, 32);
        if (m == 0) {
            int tok = t0 + r + 8 * h;
            out[tok] = g * v + xmean[tok];
        }
    }
}

extern "C" void kernel_launch(void* const* d_in, const int* in_sizes, int n_in,
                              void* d_out, int out_size, void* d_ws, size_t ws_size,
                              hipStream_t stream) {
    const float* x     = (const float*)d_in[0];
    const float* Wq    = (const float*)d_in[1];
    const float* bq    = (const float*)d_in[2];
    const float* Wk    = (const float*)d_in[3];
    const float* bk    = (const float*)d_in[4];
    const float* Wv    = (const float*)d_in[5];
    const float* bv    = (const float*)d_in[6];
    const float* gamma = (const float*)d_in[7];
    float* out = (float*)d_out;

    // workspace layout (elements are ushort unless noted), ~36.3 MB total
    unsigned short* Qb  = (unsigned short*)d_ws;
    unsigned short* Kb  = Qb  + (size_t)8 * NTOK * CQK;     // +2 MB
    unsigned short* Vt  = Kb  + (size_t)8 * NTOK * CQK;     // +2 MB
    unsigned short* xb  = Vt  + (size_t)8 * CIN * NTOK;     // +16 MB
    unsigned short* Wqt = xb  + (size_t)8 * NTOK * CIN;     // +16 MB
    unsigned short* Wkt = Wqt + (size_t)CQK * CIN;
    unsigned short* Wvt = Wkt + (size_t)CQK * CIN;
    float* xmean        = (float*)(Wvt + (size_t)CIN * CIN);

    cvt_x_kernel<<<4096, 256, 0, stream>>>(x, xb, xmean);
    cvt_w_kernel<<<256, 256, 0, stream>>>(Wq, Wk, Wv, Wqt, Wkt, Wvt);
    proj_kernel<<<256, 256, 0, stream>>>(xb, Wqt, bq, Wkt, bk, Wvt, bv,
                                         Qb, Kb, Vt);
    attn_kernel<<<256, 256, 0, stream>>>(Qb, Kb, Vt, xmean, gamma, out);
}
